// Mamba_Encoder_2302102470700
// MI455X (gfx1250) — matrix operand order repaired and checked by reading, not compiled, not run on hardware
//
#include <hip/hip_runtime.h>
#include <hip/hip_bf16.h>
#include <math.h>

typedef __attribute__((ext_vector_type(16))) _Float16 v16h;
typedef __attribute__((ext_vector_type(8)))  _Float16 v8h;
typedef __attribute__((ext_vector_type(8)))  float    v8f;

#define M_TOK   32768   // 32 sequences * 1024 tokens
#define T_LEN   1024
#define NSEQ    32
#define DMODEL  128
#define DINNER  256
#define DFF     256
#define DSTATE  16
#define DTRANK  8

// ---------------------------------------------------------------------------
// helpers
// ---------------------------------------------------------------------------
__device__ __forceinline__ float gelu_exact(float v) {
    return 0.5f * v * (1.0f + erff(v * 0.70710678118654752440f));
}
__device__ __forceinline__ float silu(float v) {
    return v / (1.0f + __expf(-v));
}

// ---------------------------------------------------------------------------
// f32 -> f16 conversion (weights / activations)
// ---------------------------------------------------------------------------
__global__ void k_f32_to_f16(const float* __restrict__ src, _Float16* __restrict__ dst, int n) {
    int i = blockIdx.x * blockDim.x + threadIdx.x;
    if (i < n) dst[i] = (_Float16)src[i];
}

// ---------------------------------------------------------------------------
// patch embedding: h[seq,t,d] = sum_p x[seq,p,t] * Wp[d,p] + bp[d]
// x flat: ((seq*16)+p)*1024 + t ; writes f32 and f16 copies
// ---------------------------------------------------------------------------
__global__ void k_patch_embed(const float* __restrict__ x, const float* __restrict__ wp,
                              const float* __restrict__ bp,
                              float* __restrict__ h32, _Float16* __restrict__ h16) {
    int idx   = blockIdx.x * 256 + threadIdx.x;      // M_TOK*128 total
    int token = idx >> 7;
    int d     = idx & 127;
    int seq   = token >> 10;
    int t     = token & (T_LEN - 1);
    const float* xp = x + ((size_t)seq * 16) * T_LEN + t;
    float acc = bp[d];
#pragma unroll
    for (int p = 0; p < 16; ++p) acc += xp[(size_t)p * T_LEN] * wp[d * 16 + p];
    h32[idx] = acc;
    h16[idx] = (_Float16)acc;
}

// ---------------------------------------------------------------------------
// WMMA GEMM: out[M,N] = A[M,K] (f16, row-major) @ W[N,K]^T (f16, row-major)
// One wave per 16(M) x 64(N) output tile: one A fragment is reused across
// four B fragments / four accumulators per K-step (4x less A traffic,
// 4 back-to-back v_wmma per K-step). K multiple of 32, N multiple of 64.
// Optional bias[N], optional exact-GELU, optional f32 and/or f16 outputs.
// ---------------------------------------------------------------------------
__global__ void k_wmma_gemm(const _Float16* __restrict__ Amat,
                            const _Float16* __restrict__ Wmat,
                            const float* __restrict__ bias,
                            float* __restrict__ outF32,
                            _Float16* __restrict__ outF16,
                            int M, int N, int K, int act) {
    const int lane  = threadIdx.x & 31;
    const int tileM = blockIdx.x << 4;
    const int tileN = blockIdx.y << 6;   // 64 columns per wave
    const int mr    = lane & 15;
    const int khi   = lane >> 4;         // K-half select

    // A frag: lane(l<16) row m=l holds K {0..15}; lane(l+16) holds {16..31}
    const _Float16* Arow = Amat + (size_t)(tileM + mr) * K + khi * 16;
    // B frag j: lane holds 16 consecutive K of weight row (tileN + 16j + mr)
    const _Float16* Wrow = Wmat + (size_t)(tileN + mr) * K + khi * 16;
    const size_t wj = (size_t)16 * K;    // stride between 16-column subtiles

    v8f acc0 = {}, acc1 = {}, acc2 = {}, acc3 = {};
    for (int k0 = 0; k0 < K; k0 += 32) {
        __builtin_prefetch((const void*)(Arow + k0 + 128), 0, 0);
        __builtin_prefetch((const void*)(Wrow + k0 + 128), 0, 0);
        v8h alo = *(const v8h*)(Arow + k0);
        v8h ahi = *(const v8h*)(Arow + k0 + 8);
        v16h af = __builtin_shufflevector(alo, ahi, 0, 1, 2, 3, 4, 5, 6, 7,
                                                    8, 9, 10, 11, 12, 13, 14, 15);
#define LOAD_B(j)                                                             \
        v8h blo##j = *(const v8h*)(Wrow + (size_t)(j) * wj + k0);             \
        v8h bhi##j = *(const v8h*)(Wrow + (size_t)(j) * wj + k0 + 8);         \
        v16h bf##j = __builtin_shufflevector(blo##j, bhi##j, 0, 1, 2, 3, 4, 5, 6, 7, \
                                                             8, 9, 10, 11, 12, 13, 14, 15)
        LOAD_B(0); LOAD_B(1); LOAD_B(2); LOAD_B(3);
#undef LOAD_B
        acc0 = __builtin_amdgcn_wmma_f32_16x16x32_f16(false, af, false, bf0,
                                                      (short)0, acc0, false, false);
        acc1 = __builtin_amdgcn_wmma_f32_16x16x32_f16(false, af, false, bf1,
                                                      (short)0, acc1, false, false);
        acc2 = __builtin_amdgcn_wmma_f32_16x16x32_f16(false, af, false, bf2,
                                                      (short)0, acc2, false, false);
        acc3 = __builtin_amdgcn_wmma_f32_16x16x32_f16(false, af, false, bf3,
                                                      (short)0, acc3, false, false);
    }

#pragma unroll
    for (int j = 0; j < 4; ++j) {
        const v8f acc = (j == 0) ? acc0 : (j == 1) ? acc1 : (j == 2) ? acc2 : acc3;
        const int   n  = tileN + j * 16 + mr;
        const float bn = bias ? bias[n] : 0.0f;
#pragma unroll
        for (int r = 0; r < 8; ++r) {
            int m   = tileM + r + (khi << 3);
            float v = acc[r] + bn;
            if (act == 1) v = gelu_exact(v);
            size_t o = (size_t)m * N + n;
            if (outF32) outF32[o] = v;
            if (outF16) outF16[o] = (_Float16)v;
        }
    }
}

// ---------------------------------------------------------------------------
// depthwise causal conv(4) + SiLU over xin = xz[:, 0:256]
// ---------------------------------------------------------------------------
__global__ void k_conv_silu(const float* __restrict__ xz, const float* __restrict__ cw,
                            const float* __restrict__ cb, float* __restrict__ xc) {
    int idx   = blockIdx.x * 256 + threadIdx.x;      // M_TOK*256
    int token = idx >> 8;
    int d     = idx & 255;
    int t     = token & (T_LEN - 1);
    const float* base = xz + (size_t)token * (2 * DINNER) + d;
    float acc = cb[d];
#pragma unroll
    for (int k = 0; k < 4; ++k) {
        int off = k - 3;                             // taps at t-3 .. t
        if (t + off >= 0) acc += cw[d * 4 + k] * base[(long)off * (2 * DINNER)];
    }
    xc[idx] = silu(acc);
}

// ---------------------------------------------------------------------------
// x_proj: x_dbl[token, o<40] = xc[token,:] . xpw[o,:]   (skinny N -> VALU)
// ---------------------------------------------------------------------------
__global__ void k_xproj(const float* __restrict__ xc, const float* __restrict__ xpw,
                        float* __restrict__ xdbl) {
    int idx   = blockIdx.x * 256 + threadIdx.x;      // M_TOK*40
    int token = idx / 40;
    int o     = idx - token * 40;
    const float4* a4 = (const float4*)(xc + (size_t)token * DINNER);
    const float4* w4 = (const float4*)(xpw + (size_t)o * DINNER);
    float acc = 0.f;
#pragma unroll 8
    for (int i = 0; i < DINNER / 4; ++i) {
        float4 a = a4[i], w = w4[i];
        acc += a.x * w.x + a.y * w.y + a.z * w.z + a.w * w.w;
    }
    xdbl[idx] = acc;
}

// ---------------------------------------------------------------------------
// dt_proj + softplus: delta[token,d] = softplus(dt . dtw[d,:] + dtb[d])
// ---------------------------------------------------------------------------
__global__ void k_dtproj(const float* __restrict__ xdbl, const float* __restrict__ dtw,
                         const float* __restrict__ dtb, float* __restrict__ delta) {
    int idx   = blockIdx.x * 256 + threadIdx.x;      // M_TOK*256
    int token = idx >> 8;
    int d     = idx & 255;
    const float* dt = xdbl + (size_t)token * 40;
    float acc = dtb[d];
#pragma unroll
    for (int r = 0; r < DTRANK; ++r) acc += dt[r] * dtw[d * DTRANK + r];
    delta[idx] = (acc > 20.f) ? acc : log1pf(__expf(acc));
}

// ---------------------------------------------------------------------------
// selective scan, fused with D-skip and SiLU(z) gating; writes f16 for out_proj.
// One block per sequence, one thread per channel; 16-state in VGPRs.
// Per-token B/C broadcast through double-buffered LDS: one barrier per step.
// exp() folded to native base-2: a2 = A * log2(e), h = exp2(dl*a2)*h + du*B.
// ---------------------------------------------------------------------------
__global__ void k_scan(const float* __restrict__ xc, const float* __restrict__ delta,
                       const float* __restrict__ xdbl, const float* __restrict__ xz,
                       const float* __restrict__ A_log, const float* __restrict__ Dp,
                       _Float16* __restrict__ yg) {
    __shared__ float sB[2][DSTATE];
    __shared__ float sC[2][DSTATE];
    const int seq = blockIdx.x;
    const int d   = threadIdx.x;                     // 0..255
    float a2[DSTATE], h[DSTATE];
#pragma unroll
    for (int s = 0; s < DSTATE; ++s) {
        a2[s] = -__expf(A_log[d * DSTATE + s]) * 1.44269504088896340736f;
        h[s]  = 0.f;
    }
    const float Dd  = Dp[d];
    const size_t b0 = (size_t)seq * T_LEN;

    // preload t = 0 into buffer 0
    if (threadIdx.x < DSTATE)
        sB[0][threadIdx.x] = xdbl[b0 * 40 + DTRANK + threadIdx.x];
    else if (threadIdx.x < 2 * DSTATE)
        sC[0][threadIdx.x - DSTATE] = xdbl[b0 * 40 + DTRANK + DSTATE + (threadIdx.x - DSTATE)];
    __syncthreads();

    for (int t = 0; t < T_LEN; ++t) {
        const int p = t & 1;
        if (t + 1 < T_LEN) {                         // stage next token's B/C
            const size_t nt = b0 + t + 1;
            if (threadIdx.x < DSTATE)
                sB[p ^ 1][threadIdx.x] = xdbl[nt * 40 + DTRANK + threadIdx.x];
            else if (threadIdx.x < 2 * DSTATE)
                sC[p ^ 1][threadIdx.x - DSTATE] =
                    xdbl[nt * 40 + DTRANK + DSTATE + (threadIdx.x - DSTATE)];
        }
        const size_t tok = b0 + t;
        const float u  = xc[tok * DINNER + d];
        const float dl = delta[tok * DINNER + d];
        const float du = dl * u;
        float y = 0.f;
#pragma unroll
        for (int s = 0; s < DSTATE; ++s) {
            h[s] = exp2f(dl * a2[s]) * h[s] + du * sB[p][s];
            y += h[s] * sC[p][s];
        }
        y += u * Dd;
        const float z = xz[tok * (2 * DINNER) + DINNER + d];
        yg[tok * DINNER + d] = (_Float16)(y * silu(z));
        __syncthreads();
    }
}

// ---------------------------------------------------------------------------
// final layout: out[seq, d, t] = h[seq, t, d]   (out shape (4,8,128,1024))
// ---------------------------------------------------------------------------
__global__ void k_write_out(const float* __restrict__ h, float* __restrict__ out) {
    int idx = blockIdx.x * 256 + threadIdx.x;        // M_TOK*128
    int t   = idx & (T_LEN - 1);
    int d   = (idx >> 10) & 127;
    int seq = idx >> 17;
    out[idx] = h[((size_t)seq * T_LEN + t) * DMODEL + d];
}

// ---------------------------------------------------------------------------
// launcher
// ---------------------------------------------------------------------------
extern "C" void kernel_launch(void* const* d_in, const int* in_sizes, int n_in,
                              void* d_out, int out_size, void* d_ws, size_t ws_size,
                              hipStream_t stream) {
    const float* x        = (const float*)d_in[0];
    const float* W_P_w    = (const float*)d_in[1];
    const float* W_P_b    = (const float*)d_in[2];
    const float* in_proj  = (const float*)d_in[3];
    const float* conv_w   = (const float*)d_in[4];
    const float* conv_b   = (const float*)d_in[5];
    const float* x_proj   = (const float*)d_in[6];
    const float* dt_projw = (const float*)d_in[7];
    const float* dt_projb = (const float*)d_in[8];
    const float* A_log    = (const float*)d_in[9];
    const float* Dparam   = (const float*)d_in[10];
    const float* out_proj = (const float*)d_in[11];
    const float* lin1_w   = (const float*)d_in[12];
    const float* lin1_b   = (const float*)d_in[13];
    const float* lin2_w   = (const float*)d_in[14];
    const float* lin2_b   = (const float*)d_in[15];
    float* out            = (float*)d_out;

    const size_t M = M_TOK;
    char* wsp = (char*)d_ws;
    auto alloc = [&](size_t bytes) -> char* {
        char* p = wsp;
        wsp += (bytes + 255) & ~(size_t)255;
        return p;
    };
    float*    hA   = (float*)   alloc(M * DMODEL * 4);       // running f32 hidden
    _Float16* hB   = (_Float16*)alloc(M * DMODEL * 2);       // f16 hidden (GEMM input)
    float*    xz   = (float*)   alloc(M * 2 * DINNER * 4);   // in_proj output
    float*    xc   = (float*)   alloc(M * DINNER * 4);       // conv+silu output
    float*    xdbl = (float*)   alloc(M * 40 * 4);           // x_proj output
    float*    dlt  = (float*)   alloc(M * DINNER * 4);       // softplus(dt)
    _Float16* yg   = (_Float16*)alloc(M * DINNER * 2);       // gated scan output (f16)
    _Float16* mid  = (_Float16*)alloc(M * DFF * 2);          // gelu(lin1) (f16)
    _Float16* w_in = (_Float16*)alloc((size_t)2 * 512 * 128 * 2);
    _Float16* w_op = (_Float16*)alloc((size_t)2 * 128 * 256 * 2);
    _Float16* w_l1 = (_Float16*)alloc((size_t)2 * 256 * 128 * 2);
    _Float16* w_l2 = (_Float16*)alloc((size_t)2 * 128 * 256 * 2);

    // weight conversions (both layers at once)
    k_f32_to_f16<<<(2 * 512 * 128 + 255) / 256, 256, 0, stream>>>(in_proj, w_in, 2 * 512 * 128);
    k_f32_to_f16<<<(2 * 128 * 256 + 255) / 256, 256, 0, stream>>>(out_proj, w_op, 2 * 128 * 256);
    k_f32_to_f16<<<(2 * 256 * 128 + 255) / 256, 256, 0, stream>>>(lin1_w, w_l1, 2 * 256 * 128);
    k_f32_to_f16<<<(2 * 128 * 256 + 255) / 256, 256, 0, stream>>>(lin2_w, w_l2, 2 * 128 * 256);

    k_patch_embed<<<(M * DMODEL) / 256, 256, 0, stream>>>(x, W_P_w, W_P_b, hA, hB);

    for (int l = 0; l < 2; ++l) {
        // in_proj: (M,128) @ (512,128)^T -> xz (M,512)
        k_wmma_gemm<<<dim3(M / 16, 512 / 64), 32, 0, stream>>>(
            hB, w_in + (size_t)l * 512 * 128, nullptr, xz, nullptr, M, 512, 128, 0);

        k_conv_silu<<<(M * DINNER) / 256, 256, 0, stream>>>(
            xz, conv_w + (size_t)l * DINNER * 4, conv_b + (size_t)l * DINNER, xc);

        k_xproj<<<(M * 40) / 256, 256, 0, stream>>>(
            xc, x_proj + (size_t)l * 40 * DINNER, xdbl);

        k_dtproj<<<(M * DINNER) / 256, 256, 0, stream>>>(
            xdbl, dt_projw + (size_t)l * DINNER * DTRANK, dt_projb + (size_t)l * DINNER, dlt);

        k_scan<<<NSEQ, DINNER, 0, stream>>>(
            xc, dlt, xdbl, xz, A_log + (size_t)l * DINNER * DSTATE,
            Dparam + (size_t)l * DINNER, yg);

        // out_proj: (M,256) @ (128,256)^T -> hB (f16, feeds lin1)
        k_wmma_gemm<<<dim3(M / 16, 128 / 64), 32, 0, stream>>>(
            yg, w_op + (size_t)l * 128 * 256, nullptr, nullptr, hB, M, 128, 256, 0);

        // lin1 + exact gelu: (M,128) @ (256,128)^T -> mid (f16)
        k_wmma_gemm<<<dim3(M / 16, 256 / 64), 32, 0, stream>>>(
            hB, w_l1 + (size_t)l * 256 * 128, lin1_b + (size_t)l * DFF, nullptr, mid,
            M, 256, 128, 1);

        // lin2: (M,256) @ (128,256)^T -> hA (f32, final/next) + hB (f16, next in_proj)
        k_wmma_gemm<<<dim3(M / 16, 128 / 64), 32, 0, stream>>>(
            mid, w_l2 + (size_t)l * 128 * 256, lin2_b + (size_t)l * DMODEL, hA, hB,
            M, 128, 256, 0);
    }

    k_write_out<<<(M * DMODEL) / 256, 256, 0, stream>>>(hA, out);
}
